// HybridEncoderV2_88081189306785
// MI455X (gfx1250) — matrix operand order repaired and checked by reading, not compiled, not run on hardware
//
#include <hip/hip_runtime.h>

typedef unsigned short u16;
typedef __attribute__((ext_vector_type(16))) __bf16 v16bf;
typedef __attribute__((ext_vector_type(8)))  float  v8f;
typedef __attribute__((ext_vector_type(4)))  unsigned int u32x4;  // POD 16B vector (not HIP's class uint4)

union FragB { v16bf v; u32x4 q[2]; };

// ---------------- global-address-space load helpers ----------------
// Force global_load_* (not flat_load_*) even when addrspace inference fails
// (e.g. pointer selects / divergent gathers). FLAT would tie up DScnt + LDS pipe.
__device__ __forceinline__ u32x4 gld4(const u16* p) {
    return *(const __attribute__((address_space(1))) u32x4*)(unsigned long long)p;
}
__device__ __forceinline__ float gldf(const float* p) {
    return *(const __attribute__((address_space(1))) float*)(unsigned long long)p;
}
__device__ __forceinline__ u16 gldh(const u16* p) {
    return *(const __attribute__((address_space(1))) u16*)(unsigned long long)p;
}

// ---------------- numeric helpers ----------------
__device__ __forceinline__ u16 f2bf(float f) {
    unsigned int u = __float_as_uint(f);
    unsigned int r = u + 0x7FFFu + ((u >> 16) & 1u);
    return (u16)(r >> 16);
}
__device__ __forceinline__ float bf2f(u16 h) {
    return __uint_as_float(((unsigned int)h) << 16);
}

// activation ids: 0 none, 1 relu, 2 exact-gelu, 3 elu+1
template<int ACT> __device__ __forceinline__ float actf(float v) {
    if (ACT == 1) return fmaxf(v, 0.f);
    if (ACT == 2) return 0.5f * v * (1.f + erff(v * 0.70710678118654752f));
    if (ACT == 3) return v > 0.f ? v + 1.f : __expf(v);
    return v;
}

// ---------------- WMMA GEMM:  C[M,N] = act(A[M,K]bf16 @ W[N,K]bf16^T + bias) ----------------
// block = 128 threads (4 waves). wave -> one 16x16 D tile. grid = (N/16, M/64).
template<int ACT>
__global__ __launch_bounds__(128) void wmma_gemm_kernel(
    const u16* __restrict__ A, const u16* __restrict__ W,
    const float* __restrict__ bias, float* __restrict__ C,
    int lda, int K, int ldc)
{
    const int lane = threadIdx.x & 31;
    const int wave = threadIdx.x >> 5;
    const int nt = blockIdx.x;
    const int mt = blockIdx.y * 4 + wave;
    // A fragment: 16-bit A 16x32 layout (lanes 0-15: K base 0, lanes 16-31: K base 8;
    //             chunk0 = K+0..7, chunk1 = K+16..23)
    const u16* arow = A + (size_t)(mt * 16 + (lane & 15)) * lda + ((lane & 16) ? 8 : 0);
    // B fragment: 16-bit B 32x16 layout (lanes 0-15: K base 0, lanes 16-31: K base 8;
    //             chunk0 = K+0..7, chunk1 = K+16..23)
    const u16* brow = W + (size_t)(nt * 16 + (lane & 15)) * K + ((lane & 16) ? 8 : 0);
    v8f acc = {0.f, 0.f, 0.f, 0.f, 0.f, 0.f, 0.f, 0.f};
#pragma unroll 2
    for (int k0 = 0; k0 < K; k0 += 32) {
        FragB a, b;
        a.q[0] = gld4(arow + k0);
        a.q[1] = gld4(arow + k0 + 16);
        b.q[0] = gld4(brow + k0);
        b.q[1] = gld4(brow + k0 + 16);
        __builtin_prefetch(arow + k0 + 128, 0, 3);   // global_prefetch_b8, speculative
        acc = __builtin_amdgcn_wmma_f32_16x16x32_bf16(false, a.v, false, b.v,
                                                      (short)0, acc, false, false);
    }
    const int col = nt * 16 + (lane & 15);
    const float bv = bias ? bias[col] : 0.f;
    const int rbase = mt * 16 + ((lane & 16) ? 8 : 0);
#pragma unroll
    for (int r = 0; r < 8; ++r)
        C[(size_t)(rbase + r) * ldc + col] = actf<ACT>(acc[r] + bv);
}

// ---------------- WMMA implicit-im2col 1D conv + fused BN + ReLU ----------------
// X bf16 [B,Tin,384]; Wp bf16 [OC, ksz*384] (tap-major). out rows m=(b,tout).
// Tap-outer loop: validity + row pointer are loop-invariant per tap; halo handled
// branchlessly via clamped address + cndmask select (no exec-mask churn, no int div).
__global__ __launch_bounds__(128) void wmma_conv_kernel(
    const u16* __restrict__ X, const u16* __restrict__ Wp,
    const float* __restrict__ g, const float* __restrict__ bb, float* __restrict__ Y,
    int Tin, int Tout, int ksz, int stride, int oc_off, int ldy)
{
    const int C = 384;
    const int K = ksz * C;
    const int lane = threadIdx.x & 31;
    const int wave = threadIdx.x >> 5;
    const int nt = blockIdx.x;
    const int mt = blockIdx.y * 4 + wave;
    const int m0 = mt * 16 + (lane & 15);
    const int bI = m0 / Tout;
    const int tout = m0 - bI * Tout;
    const int tbase = tout * stride - (ksz >> 1);
    const int kb0 = (lane & 16) ? 8 : 0;
    const u16* brow = Wp + (size_t)(nt * 16 + (lane & 15)) * K + ((lane & 16) ? 8 : 0);
    const u16* xb = X + (size_t)bI * Tin * C;
    const u32x4 z4 = {0u, 0u, 0u, 0u};
    v8f acc = {0.f, 0.f, 0.f, 0.f, 0.f, 0.f, 0.f, 0.f};
    for (int dk = 0; dk < ksz; ++dk) {
        const int t = tbase + dk;
        const bool valid = (t >= 0) && (t < Tin);
        const int tc = min(max(t, 0), Tin - 1);
        const u16* xrow = xb + (size_t)tc * C + kb0;
        const u16* wrow = brow + dk * C;
#pragma unroll 4
        for (int k0 = 0; k0 < C; k0 += 32) {
            FragB a, b;
            u32x4 a0 = gld4(xrow + k0);
            u32x4 a1 = gld4(xrow + k0 + 16);
            a.q[0] = valid ? a0 : z4;
            a.q[1] = valid ? a1 : z4;
            b.q[0] = gld4(wrow + k0);
            b.q[1] = gld4(wrow + k0 + 16);
            acc = __builtin_amdgcn_wmma_f32_16x16x32_bf16(false, a.v, false, b.v,
                                                          (short)0, acc, false, false);
        }
    }
    const int col = nt * 16 + (lane & 15);
    const float sc = g[col], bi = bb[col];
    const int rbase = mt * 16 + ((lane & 16) ? 8 : 0);
#pragma unroll
    for (int r = 0; r < 8; ++r)
        Y[(size_t)(rbase + r) * ldy + oc_off + col] = fmaxf(acc[r] * sc + bi, 0.f);
}

// ---------------- LayerNorm over 384 cols, one wave32 per row, optional residual ----------------
__global__ __launch_bounds__(128) void ln_kernel(
    const float* X, const float* R, const float* g, const float* b, float* Y)
{
    const int lane = threadIdx.x & 31;
    const size_t row = (size_t)blockIdx.x * 4 + (threadIdx.x >> 5);
    const float* xr = X + row * 384;
    float v[12];
    float s = 0.f;
#pragma unroll
    for (int i = 0; i < 12; ++i) {
        float t = xr[lane + 32 * i];
        if (R) t += R[row * 384 + lane + 32 * i];
        v[i] = t; s += t;
    }
#pragma unroll
    for (int o = 16; o; o >>= 1) s += __shfl_xor(s, o, 32);
    const float mean = s * (1.f / 384.f);
    float vs = 0.f;
#pragma unroll
    for (int i = 0; i < 12; ++i) { float d = v[i] - mean; vs += d * d; }
#pragma unroll
    for (int o = 16; o; o >>= 1) vs += __shfl_xor(vs, o, 32);
    const float inv = rsqrtf(vs * (1.f / 384.f) + 1e-5f);
#pragma unroll
    for (int i = 0; i < 12; ++i) {
        int c = lane + 32 * i;
        Y[row * 384 + c] = (v[i] - mean) * inv * g[c] + b[c];
    }
}

// ---------------- pack / repack kernels ----------------
__global__ void pack_bf16_kernel(const float* __restrict__ src, u16* __restrict__ dst,
                                 size_t rows, int K, int Kp)
{
    size_t idx = (size_t)blockIdx.x * 256 + threadIdx.x;
    if (idx >= rows * (size_t)Kp) return;
    size_t r = idx / Kp;
    int c = (int)(idx - r * Kp);
    dst[idx] = (c < K) ? f2bf(src[r * (size_t)K + c]) : (u16)0;
}

// w[OC,IC,k] -> dst[OC, k*IC] (tap-major)
__global__ void pack_convw_kernel(const float* __restrict__ w, u16* __restrict__ dst,
                                  int OC, int IC, int k)
{
    size_t idx = (size_t)blockIdx.x * 256 + threadIdx.x;
    size_t total = (size_t)OC * IC * k;
    if (idx >= total) return;
    int kic = k * IC;
    int oc = (int)(idx / kic);
    int rem = (int)(idx - (size_t)oc * kic);
    int dk = rem / IC, ic = rem - dk * IC;
    dst[idx] = f2bf(w[((size_t)oc * IC + ic) * k + dk]);
}

// whh[576,192] -> dst[192,576] transposed bf16 (coalesced in the scan)
__global__ void pack_whht_kernel(const float* __restrict__ whh, u16* __restrict__ dst)
{
    int idx = blockIdx.x * 256 + threadIdx.x;
    if (idx >= 192 * 576) return;
    int i = idx / 576, j = idx - i * 576;
    dst[idx] = f2bf(whh[(size_t)j * 192 + i]);
}

// ---------------- small elementwise / pooling kernels ----------------
__global__ void meanT_kernel(const float* __restrict__ Y, float* __restrict__ S, int T)
{
    int idx = blockIdx.x * 256 + threadIdx.x;  // B*384 threads
    int b = idx / 384, c = idx - b * 384;
    const float* p = Y + ((size_t)b * T) * 384 + c;
    float s = 0.f;
    for (int t = 0; t < T; ++t) s += p[(size_t)t * 384];
    S[idx] = s / (float)T;
}

__global__ __launch_bounds__(384) void se_mlp_kernel(
    const float* __restrict__ S, const float* __restrict__ w1, const float* __restrict__ b1,
    const float* __restrict__ w2, const float* __restrict__ b2, float* __restrict__ SC)
{
    __shared__ float s[384];
    __shared__ float s1[96];
    int b = blockIdx.x, j = threadIdx.x;
    s[j] = S[b * 384 + j];
    __syncthreads();
    if (j < 96) {
        float a = b1[j];
        const float* wr = w1 + (size_t)j * 384;
        for (int i = 0; i < 384; ++i) a += s[i] * wr[i];
        s1[j] = fmaxf(a, 0.f);
    }
    __syncthreads();
    float a = b2[j];
    const float* wr = w2 + (size_t)j * 96;
    for (int i = 0; i < 96; ++i) a += s1[i] * wr[i];
    SC[b * 384 + j] = 1.f / (1.f + __expf(-a));
}

__global__ void scale_kernel(float* __restrict__ Y, const float* __restrict__ S, int T)
{
    size_t idx = (size_t)blockIdx.x * 256 + threadIdx.x;
    size_t row = idx / 384;
    int c = (int)(idx - row * 384);
    int b = (int)(row / T);
    Y[idx] *= S[b * 384 + c];
}

__global__ void posenc_kernel(float* __restrict__ H, int T)
{
    size_t idx = (size_t)blockIdx.x * 256 + threadIdx.x;
    size_t row = idx / 384;
    int c = (int)(idx - row * 384);
    int t = (int)(row % T);
    float f = __expf(-(float)(c & ~1) * (9.210340371976184f / 384.f));
    float ang = (float)t * f;
    H[idx] += (c & 1) ? __cosf(ang) : __sinf(ang);
}

// ---------------- GRU scan: one block per (batch, direction), 192 threads ----------------
__global__ __launch_bounds__(192) void gru_scan_kernel(
    const float* __restrict__ XGf, const float* __restrict__ XGb,
    const u16* __restrict__ WTf, const u16* __restrict__ WTb,
    const float* __restrict__ BHf, const float* __restrict__ BHb,
    float* __restrict__ Y, int T)
{
    __shared__ float h_s[192];
    const int b = blockIdx.x;
    const int dir = blockIdx.y;
    const int j = threadIdx.x;
    const float* xg = (dir ? XGb : XGf) + (size_t)b * T * 576;
    const u16* wt = dir ? WTb : WTf;   // [192][576] transposed, gate-major cols
    const float* bh = dir ? BHb : BHf;
    const float br = gldf(bh + j), bz = gldf(bh + 192 + j), bn = gldf(bh + 384 + j);
    h_s[j] = 0.f;
    __syncthreads();
    for (int s = 0; s < T; ++s) {
        const int t = dir ? (T - 1 - s) : s;
        const float* xr = xg + (size_t)t * 576;
        float ar = 0.f, az = 0.f, an = 0.f;
#pragma unroll 4
        for (int i = 0; i < 192; ++i) {
            float hi = h_s[i];
            const u16* wr = wt + i * 576 + j;
            ar += hi * bf2f(gldh(wr));
            az += hi * bf2f(gldh(wr + 192));
            an += hi * bf2f(gldh(wr + 384));
        }
        float hold = h_s[j];
        float r = 1.f / (1.f + __expf(-(gldf(xr + j) + ar + br)));
        float z = 1.f / (1.f + __expf(-(gldf(xr + 192 + j) + az + bz)));
        float n = tanhf(gldf(xr + 384 + j) + r * (an + bn));
        float hn = (1.f - z) * n + z * hold;
        __syncthreads();
        h_s[j] = hn;
        Y[((size_t)b * T + t) * 384 + dir * 192 + j] = hn;
        __syncthreads();
    }
}

// ---------------- linear attention: KV = K^T V and Ksum, per (b,head) block ----------------
__global__ __launch_bounds__(256) void attn_kv_kernel(
    const float* __restrict__ Kb, const float* __restrict__ Vb,
    float* __restrict__ KV, float* __restrict__ Ks, int T)
{
    __shared__ float kt[32][96];
    __shared__ float vt[32][96];
    __shared__ float kva[96 * 96];
    __shared__ float ksa[96];
    const int bh = blockIdx.x;
    const int b = bh >> 2, hh = bh & 3;
    const int tid = threadIdx.x;
    for (int p = tid; p < 9216; p += 256) kva[p] = 0.f;
    if (tid < 96) ksa[tid] = 0.f;
    __syncthreads();
    for (int n0 = 0; n0 < T; n0 += 32) {
        for (int idx = tid; idx < 32 * 96; idx += 256) {
            int nn = idx / 96, c = idx - nn * 96;
            size_t base = ((size_t)b * T + n0 + nn) * 384 + hh * 96 + c;
            kt[nn][c] = Kb[base];
            vt[nn][c] = Vb[base];
        }
        __syncthreads();
        for (int p = tid; p < 9216; p += 256) {
            int d = p / 96, v = p - d * 96;
            float a = kva[p];
#pragma unroll 8
            for (int nn = 0; nn < 32; ++nn) a += kt[nn][d] * vt[nn][v];
            kva[p] = a;
        }
        if (tid < 96) {
            float a = ksa[tid];
            for (int nn = 0; nn < 32; ++nn) a += kt[nn][tid];
            ksa[tid] = a;
        }
        __syncthreads();
    }
    for (int p = tid; p < 9216; p += 256) KV[(size_t)bh * 9216 + p] = kva[p];
    if (tid < 96) Ks[bh * 96 + tid] = ksa[tid];
}

// out[b,t,h*96+v] = (Q . KV) / max(Q . Ksum, 1e-4); one block per (b,t)
__global__ __launch_bounds__(384) void attn_out_kernel(
    const float* __restrict__ Q, const float* __restrict__ KV,
    const float* __restrict__ Ks, float* __restrict__ Ctx, int T)
{
    __shared__ float q[384];
    size_t row = blockIdx.x;
    int b = (int)(row / T);
    int c = threadIdx.x;
    q[c] = Q[row * 384 + c];
    __syncthreads();
    int hh = c / 96, v = c - hh * 96;
    const float* kv = KV + ((size_t)(b * 4 + hh)) * 9216;
    const float* ks = Ks + (b * 4 + hh) * 96;
    const float* qh = q + hh * 96;
    float o = 0.f, z = 0.f;
    for (int d = 0; d < 96; ++d) { float qd = qh[d]; o += qd * kv[d * 96 + v]; z += qd * ks[d]; }
    Ctx[row * 384 + c] = o / fmaxf(z, 1e-4f);
}

// enc_hidden = tanh(S @ dec_w^T + dec_b); one block per batch row
__global__ __launch_bounds__(384) void dec_kernel(
    const float* __restrict__ S, const float* __restrict__ W,
    const float* __restrict__ Bb, float* __restrict__ O)
{
    __shared__ float s[384];
    int b = blockIdx.x, j = threadIdx.x;
    s[j] = S[b * 384 + j];
    __syncthreads();
    float a = Bb[j];
    const float* wr = W + (size_t)j * 384;
    for (int i = 0; i < 384; ++i) a += s[i] * wr[i];
    O[b * 384 + j] = tanhf(a);
}

// ---------------- host-side helpers ----------------
static void launch_gemm(int act, const u16* A, const u16* W, const float* bias, float* C,
                        int lda, int K, int N, size_t M, int ldc, hipStream_t s)
{
    dim3 g((unsigned)(N / 16), (unsigned)(M / 64)), b(128);
    switch (act) {
        case 2: wmma_gemm_kernel<2><<<g, b, 0, s>>>(A, W, bias, C, lda, K, ldc); break;
        case 3: wmma_gemm_kernel<3><<<g, b, 0, s>>>(A, W, bias, C, lda, K, ldc); break;
        default: wmma_gemm_kernel<0><<<g, b, 0, s>>>(A, W, bias, C, lda, K, ldc); break;
    }
}

static void launch_pack(const float* src, u16* dst, size_t rows, int K, int Kp, hipStream_t s)
{
    size_t total = rows * (size_t)Kp;
    pack_bf16_kernel<<<(unsigned)((total + 255) / 256), 256, 0, s>>>(src, dst, rows, K, Kp);
}

extern "C" void kernel_launch(void* const* d_in, const int* in_sizes, int n_in,
                              void* d_out, int out_size, void* d_ws, size_t ws_size,
                              hipStream_t stream)
{
    (void)in_sizes; (void)n_in; (void)out_size; (void)ws_size;
    const int B = 32, T1 = 1024, T2 = 512, H = 384, IN = 414, INP = 416;
    const size_t M1 = (size_t)B * T1, M2 = (size_t)B * T2;
    const int ksz[4] = {3, 5, 7, 11};

    // ---- input pointer map (setup_inputs dict order) ----
    int ii = 0;
    auto in = [&](void) { return (const float*)d_in[ii++]; };
    const float* X      = in();
    const float* Wip    = in(); const float* Bip = in();
    const float* Gin    = in(); const float* Bin = in();
    const float *cw[2][4], *cg[2][4], *cb[2][4];
    const float *sew1[2], *seb1[2], *sew2[2], *seb2[2];
    for (int ci = 0; ci < 2; ++ci) {
        for (int k = 0; k < 4; ++k) { cw[ci][k] = in(); cg[ci][k] = in(); cb[ci][k] = in(); }
        sew1[ci] = in(); seb1[ci] = in(); sew2[ci] = in(); seb2[ci] = in();
    }
    const float* subw = in(); const float* subg = in(); const float* subb = in();
    const float *wih[3][2], *whh[3][2], *bih[3][2], *bhh[3][2], *gng[3], *gnb[3];
    for (int l = 0; l < 3; ++l) {
        for (int d = 0; d < 2; ++d) { wih[l][d] = in(); whh[l][d] = in(); bih[l][d] = in(); bhh[l][d] = in(); }
        gng[l] = in(); gnb[l] = in();
    }
    const float* qw = in(); const float* qb = in();
    const float* kw = in(); const float* kb = in();
    const float* vw = in(); const float* vb = in();
    const float* ow = in(); const float* ob = in();
    const float* ang = in(); const float* anb = in();
    const float* fw1 = in(); const float* fb1 = in();
    const float* fw2 = in(); const float* fb2 = in();
    const float* fng = in(); const float* fnb = in();
    const float* dw = in();  const float* db = in();

    // ---- workspace layout (bump allocator, 256B aligned) ----
    size_t cur = 0;
    auto alloc = [&](size_t bytes) {
        cur = (cur + 255) & ~(size_t)255;
        void* p = (char*)d_ws + cur;
        cur += bytes;
        return p;
    };
    u16* pw_ip = (u16*)alloc((size_t)H * INP * 2);
    u16* pw_cv[2][4];
    for (int ci = 0; ci < 2; ++ci)
        for (int k = 0; k < 4; ++k) pw_cv[ci][k] = (u16*)alloc((size_t)96 * ksz[k] * H * 2);
    u16* pw_sub = (u16*)alloc((size_t)H * 3 * H * 2);
    u16 *pw_wih[3][2], *pw_wht[3][2];
    for (int l = 0; l < 3; ++l)
        for (int d = 0; d < 2; ++d) {
            pw_wih[l][d] = (u16*)alloc((size_t)576 * 384 * 2);
            pw_wht[l][d] = (u16*)alloc((size_t)192 * 576 * 2);
        }
    u16* pw_q = (u16*)alloc((size_t)H * H * 2);
    u16* pw_k = (u16*)alloc((size_t)H * H * 2);
    u16* pw_v = (u16*)alloc((size_t)H * H * 2);
    u16* pw_o = (u16*)alloc((size_t)H * H * 2);
    u16* pw_f1 = (u16*)alloc((size_t)768 * 384 * 2);
    u16* pw_f2 = (u16*)alloc((size_t)384 * 768 * 2);

    u16*  xbf  = (u16*)alloc(M1 * INP * 2);
    float* h1  = (float*)alloc(M1 * 384 * 4);   // LN-out / conv1-out / later FF hidden (M2*768)
    float* y1  = (float*)alloc(M1 * 384 * 4);   // conv0-out / later V + FF hidden bf16
    u16*  hbf  = (u16*)alloc(M1 * 384 * 2);     // bf16 staging of M1 activations / ctx
    float* h2  = (float*)alloc(M2 * 384 * 4);   // running hidden state at T/2
    u16*  h2bf = (u16*)alloc(M2 * 384 * 2);
    float* xgf = (float*)alloc(M2 * 576 * 4);   // fwd GRU gates / later Q / O
    float* xgb = (float*)alloc(M2 * 576 * 4);   // bwd GRU gates / later K
    float* yb  = (float*)alloc(M2 * 384 * 4);   // GRU concat / ctx / FF out
    float* kvb = (float*)alloc((size_t)128 * 9216 * 4);
    float* ksb = (float*)alloc((size_t)128 * 96 * 4);
    float* spool  = (float*)alloc((size_t)B * 384 * 4);
    float* sscale = (float*)alloc((size_t)B * 384 * 4);

    float* Vbuf  = y1;
    u16*  ff1bf  = (u16*)y1;
    float* ff1   = h1;
    float* Qb    = xgf;
    float* Kbuf  = xgb;
    float* Obuf  = xgf;
    u16*  ctxbf  = hbf;
    float* hout  = (float*)d_out;               // final h [B,T2,384]
    float* ench  = hout + M2 * 384;             // enc_hidden [B,384]

    // ---- pack weights to bf16 ----
    launch_pack(Wip, pw_ip, H, IN, INP, stream);
    for (int ci = 0; ci < 2; ++ci)
        for (int k = 0; k < 4; ++k) {
            size_t tot = (size_t)96 * H * ksz[k];
            pack_convw_kernel<<<(unsigned)((tot + 255) / 256), 256, 0, stream>>>(
                cw[ci][k], pw_cv[ci][k], 96, H, ksz[k]);
        }
    {
        size_t tot = (size_t)H * H * 3;
        pack_convw_kernel<<<(unsigned)((tot + 255) / 256), 256, 0, stream>>>(subw, pw_sub, H, H, 3);
    }
    for (int l = 0; l < 3; ++l)
        for (int d = 0; d < 2; ++d) {
            launch_pack(wih[l][d], pw_wih[l][d], 576, 384, 384, stream);
            pack_whht_kernel<<<(192 * 576 + 255) / 256, 256, 0, stream>>>(whh[l][d], pw_wht[l][d]);
        }
    launch_pack(qw, pw_q, H, H, H, stream);
    launch_pack(kw, pw_k, H, H, H, stream);
    launch_pack(vw, pw_v, H, H, H, stream);
    launch_pack(ow, pw_o, H, H, H, stream);
    launch_pack(fw1, pw_f1, 768, 384, 384, stream);
    launch_pack(fw2, pw_f2, 384, 768, 768, stream);

    // ---- input projection + LN ----
    launch_pack(X, xbf, M1, IN, INP, stream);
    launch_gemm(0, xbf, pw_ip, Bip, h1, INP, INP, H, M1, H, stream);
    ln_kernel<<<(unsigned)(M1 / 4), 128, 0, stream>>>(h1, nullptr, Gin, Bin, h1);

    // ---- multi-scale conv block 0 ----
    launch_pack(h1, hbf, M1, 384, 384, stream);
    for (int k = 0; k < 4; ++k)
        wmma_conv_kernel<<<dim3(96 / 16, (unsigned)(M1 / 64)), 128, 0, stream>>>(
            hbf, pw_cv[0][k], cg[0][k], cb[0][k], y1, T1, T1, ksz[k], 1, 96 * k, 384);
    meanT_kernel<<<(B * 384) / 256, 256, 0, stream>>>(y1, spool, T1);
    se_mlp_kernel<<<B, 384, 0, stream>>>(spool, sew1[0], seb1[0], sew2[0], seb2[0], sscale);
    scale_kernel<<<(unsigned)(M1 * 384 / 256), 256, 0, stream>>>(y1, sscale, T1);

    // ---- multi-scale conv block 1 ----
    launch_pack(y1, hbf, M1, 384, 384, stream);
    for (int k = 0; k < 4; ++k)
        wmma_conv_kernel<<<dim3(96 / 16, (unsigned)(M1 / 64)), 128, 0, stream>>>(
            hbf, pw_cv[1][k], cg[1][k], cb[1][k], h1, T1, T1, ksz[k], 1, 96 * k, 384);
    meanT_kernel<<<(B * 384) / 256, 256, 0, stream>>>(h1, spool, T1);
    se_mlp_kernel<<<B, 384, 0, stream>>>(spool, sew1[1], seb1[1], sew2[1], seb2[1], sscale);
    scale_kernel<<<(unsigned)(M1 * 384 / 256), 256, 0, stream>>>(h1, sscale, T1);

    // ---- stride-2 subsample conv + BN + ReLU, then positional encoding ----
    launch_pack(h1, hbf, M1, 384, 384, stream);
    wmma_conv_kernel<<<dim3(384 / 16, (unsigned)(M2 / 64)), 128, 0, stream>>>(
        hbf, pw_sub, subg, subb, h2, T1, T2, 3, 2, 0, 384);
    posenc_kernel<<<(unsigned)(M2 * 384 / 256), 256, 0, stream>>>(h2, T2);

    // ---- 3 bidirectional GRU layers ----
    for (int l = 0; l < 3; ++l) {
        launch_pack(h2, h2bf, M2, 384, 384, stream);
        launch_gemm(0, h2bf, pw_wih[l][0], bih[l][0], xgf, 384, 384, 576, M2, 576, stream);
        launch_gemm(0, h2bf, pw_wih[l][1], bih[l][1], xgb, 384, 384, 576, M2, 576, stream);
        gru_scan_kernel<<<dim3(B, 2), 192, 0, stream>>>(
            xgf, xgb, pw_wht[l][0], pw_wht[l][1], bhh[l][0], bhh[l][1], yb, T2);
        ln_kernel<<<(unsigned)(M2 / 4), 128, 0, stream>>>(yb, h2, gng[l], gnb[l], h2);
    }

    // ---- linear self-attention ----
    launch_pack(h2, h2bf, M2, 384, 384, stream);
    launch_gemm(3, h2bf, pw_q, qb, Qb, 384, 384, 384, M2, 384, stream);   // elu(x)+1
    launch_gemm(3, h2bf, pw_k, kb, Kbuf, 384, 384, 384, M2, 384, stream);
    launch_gemm(0, h2bf, pw_v, vb, Vbuf, 384, 384, 384, M2, 384, stream);
    attn_kv_kernel<<<128, 256, 0, stream>>>(Kbuf, Vbuf, kvb, ksb, T2);
    attn_out_kernel<<<(unsigned)M2, 384, 0, stream>>>(Qb, kvb, ksb, yb, T2);
    launch_pack(yb, ctxbf, M2, 384, 384, stream);
    launch_gemm(0, ctxbf, pw_o, ob, Obuf, 384, 384, 384, M2, 384, stream);
    ln_kernel<<<(unsigned)(M2 / 4), 128, 0, stream>>>(Obuf, h2, ang, anb, h2);

    // ---- feed-forward (exact GELU) ----
    launch_pack(h2, h2bf, M2, 384, 384, stream);
    launch_gemm(2, h2bf, pw_f1, fb1, ff1, 384, 384, 768, M2, 768, stream);
    launch_pack(ff1, ff1bf, M2, 768, 768, stream);
    launch_gemm(0, ff1bf, pw_f2, fb2, yb, 768, 768, 384, M2, 384, stream);
    ln_kernel<<<(unsigned)(M2 / 4), 128, 0, stream>>>(yb, h2, fng, fnb, hout);

    // ---- pooled decoder head ----
    meanT_kernel<<<(B * 384) / 256, 256, 0, stream>>>(hout, spool, T2);
    dec_kernel<<<B, 384, 0, stream>>>(spool, dw, db, ench);
}
